// PlanningNetworkMP_65979287601690
// MI455X (gfx1250) — matrix operand order repaired and checked by reading, not compiled, not run on hardware
//
#include <hip/hip_runtime.h>
#include <math.h>

typedef _Float16 half_t;
typedef __attribute__((ext_vector_type(16))) _Float16 v16h;
typedef __attribute__((ext_vector_type(8)))  _Float16 v8h;
typedef __attribute__((ext_vector_type(8)))  float    v8f;

#define NSEG     7
#define MTILE    128
#define NTHREADS 256

// ---- LDS byte offsets (all 16B aligned) ----
#define OFF_W3F 0        // 256x256 f16, B-fragment swizzled  (131072 B)
#define OFF_W2F 131072   // 32x256  f16, B-fragment swizzled  (16384 B)
#define OFF_WHF 147456   // 256x16  f16 packed heads, swizzled (8192 B)
#define OFF_H2  155648   // 128x256 f16 activations (reused for H3) (65536 B)
#define OFF_X1  221184   // 128x32  f16 layer-1 output         (8192 B)
#define OFF_INP 229376   // 128x4   f32 input features         (2048 B)
#define OFF_HS  231424   // 128x4   f32 head staging           (2048 B)
#define OFF_W1  233472   // 4x32    f32                        (512 B)
#define OFF_B1  233984   // 32      f32                        (128 B)
#define OFF_B2  234112   // 256     f32                        (1024 B)
#define OFF_B3  235136   // 256     f32                        (1024 B)
#define SMEM_BYTES 236160

__device__ __forceinline__ v16h pack16(v8h lo, v8h hi8) {
  v16h r;
#pragma unroll
  for (int i = 0; i < 8; ++i) { r[i] = lo[i]; r[i + 8] = hi8[i]; }
  return r;
}

// B fragment stored in swizzled order: frag base + lane*16 halves (2 x b128).
__device__ __forceinline__ v16h load_bfrag(const half_t* fragBase, int lane) {
  const half_t* p = fragBase + lane * 16;
  v8h lo = *(const v8h*)(p);
  v8h hi = *(const v8h*)(p + 8);
  return pack16(lo, hi);
}

// 16x32 f16 A fragment from row-major [rows][ldk] f16 LDS buffer.
// Lanes 0-15: row=laneLo K={0..15}; lanes 16-31: K={16..31}.
__device__ __forceinline__ v16h load_afrag(const half_t* rowPtr, int kOff, bool hi) {
  int base = kOff + (hi ? 16 : 0);
  v8h lo = *(const v8h*)(rowPtr + base);
  v8h hh = *(const v8h*)(rowPtr + base + 8);
  return pack16(lo, hh);
}

// Branch-free tanh: 1 - 2/(e^{2x}+1); saturates via inf/0 in rcp. No EXEC churn.
__device__ __forceinline__ float tanh_fast(float x) {
  float e = __expf(2.0f * x);
  return 1.0f - 2.0f * __builtin_amdgcn_rcpf(e + 1.0f);
}
__device__ __forceinline__ float sigmoid_fast(float x) {
  return __builtin_amdgcn_rcpf(1.0f + __expf(-x));
}
// Branch-free atan (minimax poly on folded argument; selects lower to cndmask).
__device__ __forceinline__ float atan_fast(float x) {
  float ax = fabsf(x);
  float hi = fmaxf(ax, 1.0f);
  float lo = fminf(ax, 1.0f);
  float t  = lo * __builtin_amdgcn_rcpf(hi);   // t in [0,1]
  float s  = t * t;
  float p = -0.0159569028f;
  p = fmaf(p, s,  0.0722072683f);
  p = fmaf(p, s, -0.1442273915f);
  p = fmaf(p, s,  0.1984969730f);
  p = fmaf(p, s, -0.3302995151f);
  p = fmaf(p, s,  0.9998329296f);
  float r = t * p;
  r = (ax > 1.0f) ? (1.57079632679f - r) : r;
  return (x < 0.0f) ? -r : r;
}

__global__ void __launch_bounds__(NTHREADS, 1)
planning_mlp_kernel(const float* __restrict__ data,
                    const float* __restrict__ W1, const float* __restrict__ b1,
                    const float* __restrict__ W2, const float* __restrict__ b2,
                    const float* __restrict__ W3, const float* __restrict__ b3,
                    const float* __restrict__ Wx,  const float* __restrict__ bx,
                    const float* __restrict__ Wy,  const float* __restrict__ by,
                    const float* __restrict__ Wdy, const float* __restrict__ bdy,
                    const float* __restrict__ Wddy,const float* __restrict__ bddy,
                    const float* __restrict__ bias_x, const float* __restrict__ bias_y,
                    const float* __restrict__ bias_dy,const float* __restrict__ bias_ddy,
                    float* __restrict__ out)
{
  extern __shared__ char smem[];
  half_t* sW3f = (half_t*)(smem + OFF_W3F);
  half_t* sW2f = (half_t*)(smem + OFF_W2F);
  half_t* sWhf = (half_t*)(smem + OFF_WHF);
  half_t* sH2  = (half_t*)(smem + OFF_H2);
  half_t* sX1  = (half_t*)(smem + OFF_X1);
  float*  sInp = (float*)(smem + OFF_INP);
  float*  sHS  = (float*)(smem + OFF_HS);
  float*  sW1  = (float*)(smem + OFF_W1);
  float*  sb1  = (float*)(smem + OFF_B1);
  float*  sb2  = (float*)(smem + OFF_B2);
  float*  sb3  = (float*)(smem + OFF_B3);

  const int tid    = threadIdx.x;
  const int lane   = tid & 31;
  const int wave   = tid >> 5;
  const int laneLo = lane & 15;
  const bool hi    = lane >= 16;

  // ---------- cooperative weight staging into swizzled-f16 LDS ----------
  for (int idx = tid; idx < 256 * 256; idx += NTHREADS) {       // W3
    int k = idx >> 8, n = idx & 255;
    int kr = k & 31;
    int frag = (k >> 5) * 16 + (n >> 4);
    int l = (n & 15) + ((kr & 16) ? 16 : 0);
    sW3f[frag * 512 + l * 16 + (kr & 15)] = (half_t)W3[idx];
  }
  for (int idx = tid; idx < 32 * 256; idx += NTHREADS) {        // W2 (single K-chunk)
    int k = idx >> 8, n = idx & 255;
    int frag = (n >> 4);
    int l = (n & 15) + ((k & 16) ? 16 : 0);
    sW2f[frag * 512 + l * 16 + (k & 15)] = (half_t)W2[idx];
  }
  for (int idx = tid; idx < 256 * 16; idx += NTHREADS) {        // packed heads 256x16
    int k = idx >> 4, n = idx & 15;
    float v = 0.0f;
    if      (n == 0) v = Wx[k];
    else if (n == 1) v = Wy[k];
    else if (n == 2) v = Wdy[k];
    else if (n == 3) v = Wddy[k];
    int kr = k & 31;
    int frag = (k >> 5);
    int l = n + ((kr & 16) ? 16 : 0);
    sWhf[frag * 512 + l * 16 + (kr & 15)] = (half_t)v;
  }
  if (tid < 128) sW1[tid] = W1[tid];
  if (tid < 32)  sb1[tid] = b1[tid];
  if (tid < 256) { sb2[tid] = b2[tid]; sb3[tid] = b3[tid]; }
  __syncthreads();

  // ---------- per-row state (replicated in both lane halves, stays uniform) ----------
  const int wrow16 = wave * 16;
  const int myrow  = blockIdx.x * MTILE + wrow16 + laneLo;
  const float* dp = data + (size_t)myrow * 6;
  float x0 = dp[0], y0 = dp[1], th0 = dp[2];
  const float xk = dp[3], yk = dp[4], thk = dp[5];
  float last_ddy = 0.0f;
  const float hbx = bx[0], hby = by[0], hbdy = bdy[0], hbddy = bddy[0];
  const float bX = bias_x[0], bY = bias_y[0], bDY = bias_dy[0], bDDY = bias_ddy[0];
  const float inv15 = 1.0f / 15.0f;
  const float inv2pi = 0.15915494309189535f; // 1/(2*pi)
  const int   rb = wrow16 + (hi ? 8 : 0);    // C/D tile row base for this lane half

  for (int seg = 0; seg < NSEG; ++seg) {
    // ---- input features ----
    if (!hi) {
      float* ip = sInp + (wrow16 + laneLo) * 4;
      ip[0] = (xk - x0) * inv15;
      ip[1] = (yk - y0) * inv15;
      ip[2] = (thk - th0) * inv2pi;
      ip[3] = last_ddy;
    }
    __syncthreads();

    // ---- layer 1: 4 -> 32, scalar VALU, f16 output in A-friendly row-major ----
    {
      const float* ip = sInp + (wrow16 + laneLo) * 4;
      const float i0 = ip[0], i1 = ip[1], i2 = ip[2], i3 = ip[3];
      const int j0 = hi ? 16 : 0;
      half_t* xrow = sX1 + (wrow16 + laneLo) * 32;
#pragma unroll
      for (int jj = 0; jj < 16; ++jj) {
        int j = j0 + jj;
        float s = sb1[j] + i0 * sW1[j] + i1 * sW1[32 + j] + i2 * sW1[64 + j] + i3 * sW1[96 + j];
        xrow[j] = (half_t)tanh_fast(s);
      }
    }
    __syncthreads();

    // ---- GEMM1: H2[16x256 per wave] = tanh(X1(16x32) @ W2(32x256) + b2) ----
    // Software-pipelined: WMMA of tile nt overlaps tanh+store of tile nt-1,
    // so the D-register consumer is a full tile away (no WMMA->VALU NOPs).
    {
      const v16h a = load_afrag(sX1 + (wrow16 + laneLo) * 32, 0, hi);
      v8f cprev;
#pragma unroll 4
      for (int nt = 0; nt < 16; ++nt) {
        const v16h b = load_bfrag(sW2f + nt * 512, lane);
        const float bias = sb2[nt * 16 + laneLo];
        v8f c;
#pragma unroll
        for (int v = 0; v < 8; ++v) c[v] = bias;
        c = __builtin_amdgcn_wmma_f32_16x16x32_f16(false, a, false, b, (short)0, c, false, false);
        if (nt > 0) {
#pragma unroll
          for (int v = 0; v < 8; ++v)
            sH2[(rb + v) * 256 + (nt - 1) * 16 + laneLo] = (half_t)tanh_fast(cprev[v]);
        }
        cprev = c;
      }
#pragma unroll
      for (int v = 0; v < 8; ++v)
        sH2[(rb + v) * 256 + 15 * 16 + laneLo] = (half_t)tanh_fast(cprev[v]);
    }
    __syncthreads();

    // ---- GEMM2: H3 = tanh(H2(16x256) @ W3(256x256) + b3); overwrite H2 region ----
    {
      v16h afr[8];
      const half_t* ar = sH2 + (wrow16 + laneLo) * 256;
#pragma unroll
      for (int kc = 0; kc < 8; ++kc) afr[kc] = load_afrag(ar, kc * 32, hi);
      v8f cprev;
#pragma unroll 2
      for (int nt = 0; nt < 16; ++nt) {
        const float bias = sb3[nt * 16 + laneLo];
        v8f c;
#pragma unroll
        for (int v = 0; v < 8; ++v) c[v] = bias;
#pragma unroll
        for (int kc = 0; kc < 8; ++kc) {
          const v16h b = load_bfrag(sW3f + (kc * 16 + nt) * 512, lane);
          c = __builtin_amdgcn_wmma_f32_16x16x32_f16(false, afr[kc], false, b, (short)0, c, false, false);
        }
        if (nt > 0) {
#pragma unroll
          for (int v = 0; v < 8; ++v)
            sH2[(rb + v) * 256 + (nt - 1) * 16 + laneLo] = (half_t)tanh_fast(cprev[v]);
        }
        cprev = c;
      }
#pragma unroll
      for (int v = 0; v < 8; ++v)
        sH2[(rb + v) * 256 + 15 * 16 + laneLo] = (half_t)tanh_fast(cprev[v]);  // H3 in place
    }
    __syncthreads();

    // ---- head GEMM: Hd[16x16] = H3(16x256) @ Whead(256x16); cols 0..3 valid ----
    {
      v16h afr[8];
      const half_t* ar = sH2 + (wrow16 + laneLo) * 256;
#pragma unroll
      for (int kc = 0; kc < 8; ++kc) afr[kc] = load_afrag(ar, kc * 32, hi);
      v8f c;
#pragma unroll
      for (int v = 0; v < 8; ++v) c[v] = 0.0f;
#pragma unroll
      for (int kc = 0; kc < 8; ++kc) {
        const v16h b = load_bfrag(sWhf + kc * 512, lane);
        c = __builtin_amdgcn_wmma_f32_16x16x32_f16(false, afr[kc], false, b, (short)0, c, false, false);
      }
      if (laneLo < 4) {
        const int rb2 = hi ? 8 : 0;
#pragma unroll
        for (int v = 0; v < 8; ++v)
          sHS[(wrow16 + rb2 + v) * 4 + laneLo] = c[v];
      }
    }
    __syncthreads();

    // ---- epilogue: activations, output store, closed-form pose update ----
    // The reference fits the quintic at xk=xp and evaluates at s=xp, so
    // y_loc==yp and dY==dyp identically; the 6x6 solve cancels out.
    {
      const float* hp = sHS + (wrow16 + laneLo) * 4;
      const float xp   = 10.0f * sigmoid_fast(hp[0] + hbx) + bX;
      const float yp   = 10.0f * tanh_fast(hp[1] + hby) + bY;
      const float dyp  =         tanh_fast(hp[2] + hbdy) + bDY;
      const float ddyp =  2.0f * tanh_fast(hp[3] + hbddy) + bDDY;
      if (!hi) {
        float* op = out + (size_t)myrow * 28 + seg;   // out[row][comp][seg], comp stride 7
        op[0]  = xp;
        op[7]  = yp;
        op[14] = dyp;
        op[21] = ddyp;
      }
      const float cth = __cosf(th0), sth = __sinf(th0);
      const float nx = cth * xp - sth * yp + x0;
      const float ny = sth * xp + cth * yp + y0;
      th0 += atan_fast(dyp);
      x0 = nx; y0 = ny;
      last_ddy = ddyp;
    }
    __syncthreads();
  }
}

extern "C" void kernel_launch(void* const* d_in, const int* in_sizes, int n_in,
                              void* d_out, int out_size, void* d_ws, size_t ws_size,
                              hipStream_t stream) {
  (void)n_in; (void)d_ws; (void)ws_size; (void)out_size;
  const float* data = (const float*)d_in[0];
  const float* W1   = (const float*)d_in[1];
  const float* b1   = (const float*)d_in[2];
  const float* W2   = (const float*)d_in[3];
  const float* b2   = (const float*)d_in[4];
  const float* W3   = (const float*)d_in[5];
  const float* b3   = (const float*)d_in[6];
  const float* Wx   = (const float*)d_in[7];
  const float* bx   = (const float*)d_in[8];
  const float* Wy   = (const float*)d_in[9];
  const float* by   = (const float*)d_in[10];
  const float* Wdy  = (const float*)d_in[11];
  const float* bdy  = (const float*)d_in[12];
  const float* Wddy = (const float*)d_in[13];
  const float* bddy = (const float*)d_in[14];
  const float* bias_x  = (const float*)d_in[15];
  const float* bias_y  = (const float*)d_in[16];
  const float* bias_dy = (const float*)d_in[17];
  const float* bias_ddy= (const float*)d_in[18];
  float* out = (float*)d_out;

  const int B = in_sizes[0] / 6;           // 32768
  const int grid = (B + MTILE - 1) / MTILE;

  hipFuncSetAttribute((const void*)planning_mlp_kernel,
                      hipFuncAttributeMaxDynamicSharedMemorySize, SMEM_BYTES);
  planning_mlp_kernel<<<grid, NTHREADS, SMEM_BYTES, stream>>>(
      data, W1, b1, W2, b2, W3, b3, Wx, bx, Wy, by, Wdy, bdy, Wddy, bddy,
      bias_x, bias_y, bias_dy, bias_ddy, out);
}
